// ContrastiveLoss_66391604461719
// MI455X (gfx1250) — matrix operand order repaired and checked by reading, not compiled, not run on hardware
//
#include <hip/hip_runtime.h>
#include <hip/hip_bf16.h>

#define EMB_N 4096
#define EMB_D 768
#define STILES 128                               // EMB_N / 32 super-tiles (32x32) per dim
#define PAIR_WAVES (STILES * (STILES + 1) / 2)   // 8256 upper-tri super-tiles
#define MAIN_BLOCKS (PAIR_WAVES / 8)             // 1032 blocks of 8 waves
#define PAIR_COUNT 8386560.0f                    // N*(N-1)/2

typedef __attribute__((ext_vector_type(16))) __bf16 v16bf;
typedef __attribute__((ext_vector_type(8)))  float  v8f;

union FragU { v16bf v; uint4 u[2]; };

// ---------------- prep: fp32 -> bf16 hi/lo split + squared norms ----------------
__global__ __launch_bounds__(256) void prep_kernel(const float* __restrict__ emb,
                                                   __hip_bfloat16* __restrict__ hi,
                                                   __hip_bfloat16* __restrict__ lo,
                                                   float* __restrict__ sq)
{
    const int row = blockIdx.x;
    const float* x = emb + row * EMB_D;
    float s = 0.0f;
    for (int k = threadIdx.x; k < EMB_D; k += 256) {
        float v = x[k];
        s += v * v;
        __hip_bfloat16 h = __float2bfloat16(v);          // RNE high part
        float r = v - __bfloat162float(h);               // exact residual
        hi[row * EMB_D + k] = h;
        lo[row * EMB_D + k] = __float2bfloat16(r);
    }
    __shared__ float red[256];
    red[threadIdx.x] = s;
    __syncthreads();
    for (int off = 128; off > 0; off >>= 1) {
        if (threadIdx.x < off) red[threadIdx.x] += red[threadIdx.x + off];
        __syncthreads();
    }
    if (threadIdx.x == 0) sq[row] = red[0];
}

__global__ void zero_kernel(float* out) { out[0] = 0.0f; }

// -------- main: 32x32 register-blocked Gram tiles (2x2 WMMA) + fused loss --------
__global__ __launch_bounds__(256) void tile_loss_kernel(
    const __hip_bfloat16* __restrict__ Ahi,
    const __hip_bfloat16* __restrict__ Alo,
    const float* __restrict__ sq,
    const int*  __restrict__ labels,
    float* __restrict__ out)
{
    const int wave = threadIdx.x >> 5;
    const int lane = threadIdx.x & 31;
    const int W = blockIdx.x * 8 + wave;                 // 0 .. PAIR_WAVES-1

    // Decode (tm, tn), tm <= tn, row-major over upper triangle of STILES x STILES.
    float fb = 2.0f * STILES + 1.0f;
    int tm = (int)((fb - sqrtf(fb * fb - 8.0f * (float)W)) * 0.5f);
    if (tm < 0) tm = 0;
    if (tm >= STILES) tm = STILES - 1;
    while (tm > 0 && tm * STILES - (tm * (tm - 1)) / 2 > W) --tm;
    while ((tm + 1) * STILES - ((tm + 1) * tm) / 2 <= W) ++tm;
    const int tn = tm + (W - (tm * STILES - (tm * (tm - 1)) / 2));

    const int halve = lane >> 4;                         // 0 or 1 (lane half)
    const int l16   = lane & 15;

    // Fragment rows for the two M sub-tiles and two N sub-tiles.
    const int rA0 = tm * 32 + l16, rA1 = rA0 + 16;
    const int rB0 = tn * 32 + l16, rB1 = rB0 + 16;

    // A layout (16-bit 16x32): lane-half 0 -> K chunks [k, k+8) & [k+16, k+24)
    //                          lane-half 1 -> K chunks [k+8, k+16) & [k+24, k+32)
    // B layout (32x16): same K chunks as A: lane-half h -> [k+8h, k+8h+8) & [k+16+8h, k+24+8h)
    const __hip_bfloat16* aHi0 = Ahi + rA0 * EMB_D + halve * 8;
    const __hip_bfloat16* aLo0 = Alo + rA0 * EMB_D + halve * 8;
    const __hip_bfloat16* aHi1 = Ahi + rA1 * EMB_D + halve * 8;
    const __hip_bfloat16* aLo1 = Alo + rA1 * EMB_D + halve * 8;
    const __hip_bfloat16* bHi0 = Ahi + rB0 * EMB_D + halve * 8;
    const __hip_bfloat16* bLo0 = Alo + rB0 * EMB_D + halve * 8;
    const __hip_bfloat16* bHi1 = Ahi + rB1 * EMB_D + halve * 8;
    const __hip_bfloat16* bLo1 = Alo + rB1 * EMB_D + halve * 8;

    v8f acc00 = {}, acc01 = {}, acc10 = {}, acc11 = {};

    for (int k = 0; k < EMB_D; k += 32) {
        FragU ahi0, alo0, ahi1, alo1, bhi0, blo0, bhi1, blo1;
        ahi0.u[0] = *(const uint4*)(aHi0 + k);  ahi0.u[1] = *(const uint4*)(aHi0 + k + 16);
        alo0.u[0] = *(const uint4*)(aLo0 + k);  alo0.u[1] = *(const uint4*)(aLo0 + k + 16);
        ahi1.u[0] = *(const uint4*)(aHi1 + k);  ahi1.u[1] = *(const uint4*)(aHi1 + k + 16);
        alo1.u[0] = *(const uint4*)(aLo1 + k);  alo1.u[1] = *(const uint4*)(aLo1 + k + 16);
        bhi0.u[0] = *(const uint4*)(bHi0 + k);  bhi0.u[1] = *(const uint4*)(bHi0 + k + 16);
        blo0.u[0] = *(const uint4*)(bLo0 + k);  blo0.u[1] = *(const uint4*)(bLo0 + k + 16);
        bhi1.u[0] = *(const uint4*)(bHi1 + k);  bhi1.u[1] = *(const uint4*)(bHi1 + k + 16);
        blo1.u[0] = *(const uint4*)(bLo1 + k);  blo1.u[1] = *(const uint4*)(bLo1 + k + 16);

        // gram ~= hi*hi + hi*lo + lo*hi ; rotate across 4 independent accumulators
        acc00 = __builtin_amdgcn_wmma_f32_16x16x32_bf16(false, ahi0.v, false, bhi0.v, (short)0, acc00, false, false);
        acc01 = __builtin_amdgcn_wmma_f32_16x16x32_bf16(false, ahi0.v, false, bhi1.v, (short)0, acc01, false, false);
        acc10 = __builtin_amdgcn_wmma_f32_16x16x32_bf16(false, ahi1.v, false, bhi0.v, (short)0, acc10, false, false);
        acc11 = __builtin_amdgcn_wmma_f32_16x16x32_bf16(false, ahi1.v, false, bhi1.v, (short)0, acc11, false, false);
        acc00 = __builtin_amdgcn_wmma_f32_16x16x32_bf16(false, ahi0.v, false, blo0.v, (short)0, acc00, false, false);
        acc01 = __builtin_amdgcn_wmma_f32_16x16x32_bf16(false, ahi0.v, false, blo1.v, (short)0, acc01, false, false);
        acc10 = __builtin_amdgcn_wmma_f32_16x16x32_bf16(false, ahi1.v, false, blo0.v, (short)0, acc10, false, false);
        acc11 = __builtin_amdgcn_wmma_f32_16x16x32_bf16(false, ahi1.v, false, blo1.v, (short)0, acc11, false, false);
        acc00 = __builtin_amdgcn_wmma_f32_16x16x32_bf16(false, alo0.v, false, bhi0.v, (short)0, acc00, false, false);
        acc01 = __builtin_amdgcn_wmma_f32_16x16x32_bf16(false, alo0.v, false, bhi1.v, (short)0, acc01, false, false);
        acc10 = __builtin_amdgcn_wmma_f32_16x16x32_bf16(false, alo1.v, false, bhi0.v, (short)0, acc10, false, false);
        acc11 = __builtin_amdgcn_wmma_f32_16x16x32_bf16(false, alo1.v, false, bhi1.v, (short)0, acc11, false, false);
    }

    // C/D layout: VGPR r -> M = r + 8*halve (within sub-tile), N = l16.
    float lsum = 0.0f;
#pragma unroll
    for (int mi = 0; mi < 2; ++mi) {
#pragma unroll
        for (int ni = 0; ni < 2; ++ni) {
            const v8f acc = (mi == 0) ? (ni == 0 ? acc00 : acc01)
                                      : (ni == 0 ? acc10 : acc11);
            const int   j   = tn * 32 + ni * 16 + l16;
            const float sqj = sq[j];
            const int   lj  = labels[j];
            const int ibase = tm * 32 + mi * 16 + halve * 8;
#pragma unroll
            for (int r = 0; r < 8; ++r) {
                const int i = ibase + r;
                if (i < j) {                             // strict upper triangle
                    const float g     = acc[r];
                    const float d2    = fmaxf(sq[i] + sqj - 2.0f * g, 0.0f);
                    const float dist  = sqrtf(d2 + 1e-12f);
                    const float hinge = fmaxf(1.0f - dist, 0.0f);
                    lsum += (labels[i] == lj) ? d2 : hinge * hinge;
                }
            }
        }
    }
    // wave32 reduction
    lsum += __shfl_xor(lsum, 16, 32);
    lsum += __shfl_xor(lsum,  8, 32);
    lsum += __shfl_xor(lsum,  4, 32);
    lsum += __shfl_xor(lsum,  2, 32);
    lsum += __shfl_xor(lsum,  1, 32);
    if (lane == 0) atomicAdd(out, lsum * (1.0f / PAIR_COUNT));
}

// ---------------- host entry ----------------
extern "C" void kernel_launch(void* const* d_in, const int* in_sizes, int n_in,
                              void* d_out, int out_size, void* d_ws, size_t ws_size,
                              hipStream_t stream) {
    const float* emb    = (const float*)d_in[0];
    const int*   labels = (const int*)d_in[1];
    float*       out    = (float*)d_out;

    char* ws = (char*)d_ws;
    __hip_bfloat16* Ahi = (__hip_bfloat16*)ws;                                   // 6,291,456 B
    __hip_bfloat16* Alo = (__hip_bfloat16*)(ws + (size_t)EMB_N * EMB_D * 2);     // 6,291,456 B
    float*          sq  = (float*)(ws + (size_t)EMB_N * EMB_D * 4);              //    16,384 B

    zero_kernel<<<1, 1, 0, stream>>>(out);
    prep_kernel<<<EMB_N, 256, 0, stream>>>(emb, Ahi, Alo, sq);
    tile_loss_kernel<<<MAIN_BLOCKS, 256, 0, stream>>>(Ahi, Alo, sq, labels, out);
}
